// TransformerEncoderLayer_35588099015113
// MI455X (gfx1250) — matrix operand order repaired and checked by reading, not compiled, not run on hardware
//
#include <hip/hip_runtime.h>
#include <hip/hip_bf16.h>

#define NN 1024
#define BB 8
#define CC 256
#define KK 4
#define FFD 2048

typedef __attribute__((ext_vector_type(16))) __bf16 v16bf;
typedef __attribute__((ext_vector_type(8)))  float  v8f;

union FragU { v16bf v; unsigned int u[8]; };

// Pack two f32 -> packed bf16 pair (low = a, high = b).
#if defined(__has_builtin)
#if __has_builtin(__builtin_amdgcn_cvt_pk_bf16_f32)
#define HAVE_CVT_PK_BF16 1
#endif
#endif

#ifdef HAVE_CVT_PK_BF16
typedef __attribute__((ext_vector_type(2))) __bf16 v2bf;
__device__ __forceinline__ unsigned int pack2(float a, float b) {
    union { v2bf v; unsigned int u; } c;
    c.v = __builtin_amdgcn_cvt_pk_bf16_f32(a, b);
    return c.u;
}
#else
__device__ __forceinline__ unsigned int pack2(float a, float b) {
    unsigned int ua = __float_as_uint(a);
    ua += 0x7FFFu + ((ua >> 16) & 1u);
    unsigned int ub = __float_as_uint(b);
    ub += 0x7FFFu + ((ub >> 16) & 1u);
    return (ua >> 16) | (ub & 0xFFFF0000u);
}
#endif

__device__ __forceinline__ unsigned short f2bf(float x) {
    return (unsigned short)pack2(x, x);
}

// A-fragment k pattern (16-bit A 16x32, ISA 7.12.2): lanes0-15 K{0..7,16..23}, lanes16-31 +8
__device__ __forceinline__ int kA(int i, int lane) {
    int kb = (lane >> 4) << 3;
    return kb + ((i < 4) ? (2 * i) : (16 + 2 * (i - 4)));
}
// B-fragment k pattern (16-bit B 32x16): lanes0-15 K{0..7,16..23}, lanes16-31 +8, pairs packed
__device__ __forceinline__ int kB(int i, int lane) {
    return ((lane >> 4) << 3) + ((i < 4) ? (2 * i) : (16 + 2 * (i - 4)));
}

// A fragment from global f32 row (row-major, pairs contiguous)
__device__ __forceinline__ v16bf ldgA(const float* __restrict__ rp, int koff, int lane) {
    FragU f;
#pragma unroll
    for (int i = 0; i < 8; ++i) {
        int k = koff + kA(i, lane);
        float2 p = *(const float2*)(rp + k);
        f.u[i] = pack2(p.x, p.y);
    }
    return f.v;
}
// A fragment = (p0 + p1) * s from two global f32 rows
__device__ __forceinline__ v16bf ldgA2(const float* __restrict__ rp0, const float* __restrict__ rp1,
                                       float s, int koff, int lane) {
    FragU f;
#pragma unroll
    for (int i = 0; i < 8; ++i) {
        int k = koff + kA(i, lane);
        float2 a = *(const float2*)(rp0 + k);
        float2 c = *(const float2*)(rp1 + k);
        f.u[i] = pack2((a.x + c.x) * s, (a.y + c.y) * s);
    }
    return f.v;
}
// A fragment from LDS bf16 row (pairs contiguous)
__device__ __forceinline__ v16bf ldsA16(const unsigned short* rp, int koff, int lane) {
    FragU f;
#pragma unroll
    for (int i = 0; i < 8; ++i) f.u[i] = *(const unsigned int*)(rp + koff + kA(i, lane));
    return f.v;
}
// B fragment from LDS bf16 row (row = N index, pairs contiguous along K)
__device__ __forceinline__ v16bf ldsB(const unsigned short* rp, int koff, int lane) {
    FragU f;
#pragma unroll
    for (int i = 0; i < 8; ++i) f.u[i] = *(const unsigned int*)(rp + koff + kB(i, lane));
    return f.v;
}

__device__ __forceinline__ v8f wmma_bf16(v16bf a, v16bf b, v8f c) {
    return __builtin_amdgcn_wmma_f32_16x16x32_bf16(false, a, false, b, (short)0, c, false, false);
}

__device__ __forceinline__ float rmax16(float v) {
#pragma unroll
    for (int m = 1; m < 16; m <<= 1) v = fmaxf(v, __shfl_xor(v, m, 32));
    return v;
}
__device__ __forceinline__ float rsum16(float v) {
#pragma unroll
    for (int m = 1; m < 16; m <<= 1) v += __shfl_xor(v, m, 32);
    return v;
}

// ---------------- Kernel A: flash attention, K heads, out -> cat layout [b][n][k][c] ----
__global__ __launch_bounds__(128) void attn_kernel(
    const float* __restrict__ src, const float* __restrict__ matched,
    const float* __restrict__ pos, const float* __restrict__ mask,
    float* __restrict__ out_att)
{
    __shared__ unsigned short ks[32][264];   // K-chunk (m, c) bf16, +pos
    __shared__ unsigned short vt[256][34];   // V-chunk transposed (c, m) bf16
    __shared__ unsigned short pb[4][16][34]; // per-wave P tile (n, m) bf16

    const int tid = threadIdx.x, wave = tid >> 5, lane = tid & 31;
    const int lh = lane & 15, hi = lane >> 4;
    const int bid = blockIdx.x;
    const int qg   = bid & 15;
    const int kbid = bid >> 4;
    const int kk   = kbid >> 3;
    const int b    = kbid & 7;
    const int qbase = qg * 64 + wave * 16;

    // Q fragments (rows n = qbase + lh), pre-scaled by 1/sqrt(C)
    const int nrow = qbase + lh;
    const float* qs = src + ((size_t)nrow * BB + b) * CC;
    const float* qp = pos + ((size_t)nrow * BB + b) * CC;
    v16bf qf[8];
#pragma unroll
    for (int cs = 0; cs < 8; ++cs) qf[cs] = ldgA2(qs, qp, 0.0625f, cs * 32, lane);

    v8f z = {0.f,0.f,0.f,0.f,0.f,0.f,0.f,0.f};
    v8f acc[16];
#pragma unroll
    for (int t = 0; t < 16; ++t) acc[t] = z;
    float mrow[8], lsum[8];
#pragma unroll
    for (int r = 0; r < 8; ++r) { mrow[r] = -3.0e38f; lsum[r] = 0.f; }

    const float* mbase = mask + (size_t)b * NN * NN;

#pragma unroll 1
    for (int mc = 0; mc < 32; ++mc) {
        __syncthreads();
        const int mg0 = mc * 32;
        // stage K (+pos) and V^T as bf16; packed b32 stores for ks
        for (int idx = tid; idx < 32 * 128; idx += 128) {
            int m = idx >> 7, c2 = (idx & 127) * 2, mg = mg0 + m;
            const float* mp = matched + (((size_t)kk * NN + mg) * BB + b) * CC + c2;
            const float* pp = pos + ((size_t)mg * BB + b) * CC + c2;
            float2 mv = *(const float2*)mp;
            float2 pv = *(const float2*)pp;
            *(unsigned int*)&ks[m][c2] = pack2(mv.x + pv.x, mv.y + pv.y);
            unsigned int vp = pack2(mv.x, mv.y);
            vt[c2][m]     = (unsigned short)vp;
            vt[c2 + 1][m] = (unsigned short)(vp >> 16);
        }
        // prefetch next chunk of matched + pos into caches while we compute
        if (mc + 1 < 32) {
            int pm = tid >> 2, pq = tid & 3, mg = mg0 + 32 + pm;
            __builtin_prefetch(matched + (((size_t)kk * NN + mg) * BB + b) * CC + pq * 64, 0, 0);
            __builtin_prefetch(pos + ((size_t)mg * BB + b) * CC + pq * 64, 0, 0);
        }
        __syncthreads();

        v8f s0 = z, s1 = z;
#pragma unroll
        for (int cs = 0; cs < 8; ++cs) {
            v16bf b0 = ldsB(&ks[lh][0],      cs * 32, lane);
            v16bf b1 = ldsB(&ks[16 + lh][0], cs * 32, lane);
            s0 = wmma_bf16(qf[cs], b0, s0);
            s1 = wmma_bf16(qf[cs], b1, s1);
        }

        float sc[8];
#pragma unroll
        for (int r = 0; r < 8; ++r) {
            int n = qbase + r + 8 * hi;
            const float* mr = mbase + (size_t)n * NN + mg0;
            float v0 = s0[r] + mr[lh];
            float v1 = s1[r] + mr[16 + lh];
            float cm = rmax16(fmaxf(v0, v1));
            float nm = fmaxf(mrow[r], cm);
            float scale = __expf(mrow[r] - nm);
            mrow[r] = nm;
            float p0 = __expf(v0 - nm), p1 = __expf(v1 - nm);
            s0[r] = p0; s1[r] = p1;
            lsum[r] = lsum[r] * scale + rsum16(p0 + p1);
            sc[r] = scale;
        }
#pragma unroll
        for (int t = 0; t < 16; ++t) {
#pragma unroll
            for (int r = 0; r < 8; ++r) acc[t][r] *= sc[r];
        }
        // re-layout P: C/D layout -> LDS row-major -> A layout
#pragma unroll
        for (int r = 0; r < 8; ++r) {
            pb[wave][r + 8 * hi][lh]      = f2bf(s0[r]);
            pb[wave][r + 8 * hi][16 + lh] = f2bf(s1[r]);
        }
        asm volatile("s_wait_dscnt 0x0" ::: "memory");
        v16bf ap = ldsA16(&pb[wave][lh][0], 0, lane);
#pragma unroll
        for (int t = 0; t < 16; ++t) {
            v16bf bv = ldsB(&vt[t * 16 + lh][0], 0, lane);
            acc[t] = wmma_bf16(ap, bv, acc[t]);
        }
    }

#pragma unroll
    for (int t = 0; t < 16; ++t) {
        int c = t * 16 + lh;
#pragma unroll
        for (int r = 0; r < 8; ++r) {
            int n = qbase + r + 8 * hi;
            out_att[(((size_t)b * NN + n) * KK + kk) * CC + c] = acc[t][r] / lsum[r];
        }
    }
}

// ---------------- Kernel B: rec = cat @ W_agg + b_agg; LN(src + rec) -> (n,b,c) ----------
__global__ __launch_bounds__(128) void agg_ln_kernel(
    const float* __restrict__ cat, const float* __restrict__ Wagg,
    const float* __restrict__ bagg, const float* __restrict__ src,
    const float* __restrict__ g1, const float* __restrict__ be1,
    float* __restrict__ out_ln)
{
    __shared__ unsigned short wT[256][34];   // W_agg^T chunk (c, k) bf16
    const int tid = threadIdx.x, wave = tid >> 5, lane = tid & 31;
    const int lh = lane & 15, hi = lane >> 4;
    const int base = blockIdx.x * 64 + wave * 16;   // rows = (b,n) flattened, b major
    const float* arow = cat + (size_t)(base + lh) * (KK * CC);

    v8f z = {0.f,0.f,0.f,0.f,0.f,0.f,0.f,0.f};
    v8f acc[16];
#pragma unroll
    for (int t = 0; t < 16; ++t) acc[t] = z;

#pragma unroll 1
    for (int kc = 0; kc < 32; ++kc) {
        __syncthreads();
        for (int idx = tid; idx < 16 * 256; idx += 128) {
            int kl2 = (idx >> 8) * 2, c = idx & 255;
            float w0 = Wagg[(size_t)(kc * 32 + kl2) * CC + c];
            float w1 = Wagg[(size_t)(kc * 32 + kl2 + 1) * CC + c];
            *(unsigned int*)&wT[c][kl2] = pack2(w0, w1);
        }
        if (kc + 1 < 32) {
            int pk = tid >> 2, pq = tid & 3;
            __builtin_prefetch(Wagg + (size_t)((kc + 1) * 32 + pk) * CC + pq * 64, 0, 0);
        }
        __syncthreads();
        v16bf a = ldgA(arow, kc * 32, lane);
#pragma unroll
        for (int t = 0; t < 16; ++t) {
            v16bf bw = ldsB(&wT[t * 16 + lh][0], 0, lane);
            acc[t] = wmma_bf16(a, bw, acc[t]);
        }
    }
    // bias + residual (src is (n,b,c))
#pragma unroll
    for (int t = 0; t < 16; ++t) {
        int c = t * 16 + lh;
        float bb = bagg[c];
#pragma unroll
        for (int r = 0; r < 8; ++r) {
            int rowg = base + r + 8 * hi;
            int bi = rowg >> 10, n = rowg & 1023;
            acc[t][r] += bb + src[((size_t)n * BB + bi) * CC + c];
        }
    }
    // LayerNorm over c
    float mu[8], rs[8];
#pragma unroll
    for (int r = 0; r < 8; ++r) {
        float s = 0.f, q = 0.f;
#pragma unroll
        for (int t = 0; t < 16; ++t) { float v = acc[t][r]; s += v; q += v * v; }
        s = rsum16(s); q = rsum16(q);
        float m = s * (1.0f / 256.0f);
        float var = q * (1.0f / 256.0f) - m * m;
        mu[r] = m; rs[r] = rsqrtf(var + 1e-5f);
    }
#pragma unroll
    for (int t = 0; t < 16; ++t) {
        int c = t * 16 + lh;
        float g = g1[c], bb = be1[c];
#pragma unroll
        for (int r = 0; r < 8; ++r) {
            int rowg = base + r + 8 * hi;
            int bi = rowg >> 10, n = rowg & 1023;
            out_ln[((size_t)n * BB + bi) * CC + c] = (acc[t][r] - mu[r]) * rs[r] * g + bb;
        }
    }
}

// ---------------- Kernel C: fused FFN + residual + LN2 -> d_out (n,b,c) -----------------
__global__ __launch_bounds__(128) void ffn_ln_kernel(
    const float* __restrict__ x, const float* __restrict__ W1, const float* __restrict__ b1,
    const float* __restrict__ W2, const float* __restrict__ b2,
    const float* __restrict__ g2, const float* __restrict__ be2,
    float* __restrict__ out)
{
    __shared__ unsigned short wT[256][34];     // W1^T / W2^T chunk (n|c, k) bf16
    __shared__ unsigned short hb[4][16][136];  // per-wave relu(h) tile (n, ff) bf16
    const int tid = threadIdx.x, wave = tid >> 5, lane = tid & 31;
    const int lh = lane & 15, hi = lane >> 4;
    const int base = blockIdx.x * 64 + wave * 16;  // rows = (n,b) flattened
    const float* arow = x + (size_t)(base + lh) * CC;

    v8f z = {0.f,0.f,0.f,0.f,0.f,0.f,0.f,0.f};
    v8f oacc[16];
#pragma unroll
    for (int t = 0; t < 16; ++t) oacc[t] = z;

#pragma unroll 1
    for (int f = 0; f < 16; ++f) {           // ff chunks of 128
        v8f hacc[8];
#pragma unroll
        for (int t = 0; t < 8; ++t) hacc[t] = z;
#pragma unroll 1
        for (int kc = 0; kc < 8; ++kc) {     // K = 256
            __syncthreads();
            for (int idx = tid; idx < 16 * 128; idx += 128) {
                int kl2 = (idx >> 7) * 2, nl = idx & 127;
                float w0 = W1[(size_t)(kc * 32 + kl2) * FFD + f * 128 + nl];
                float w1 = W1[(size_t)(kc * 32 + kl2 + 1) * FFD + f * 128 + nl];
                *(unsigned int*)&wT[nl][kl2] = pack2(w0, w1);
            }
            if (kc + 1 < 8) {
                int pk = tid >> 1, pq = tid & 1;
                __builtin_prefetch(W1 + (size_t)((kc + 1) * 32 + pk) * FFD + f * 128 + pq * 64, 0, 0);
            }
            __syncthreads();
            v16bf a = ldgA(arow, kc * 32, lane);
#pragma unroll
            for (int t = 0; t < 8; ++t) {
                v16bf bw = ldsB(&wT[t * 16 + lh][0], 0, lane);
                hacc[t] = wmma_bf16(a, bw, hacc[t]);
            }
        }
        // bias + relu -> LDS (re-layout to A fragments)
#pragma unroll
        for (int t = 0; t < 8; ++t) {
            int c = f * 128 + t * 16 + lh;
            float bb = b1[c];
#pragma unroll
            for (int r = 0; r < 8; ++r) {
                float h = fmaxf(hacc[t][r] + bb, 0.0f);
                hb[wave][r + 8 * hi][t * 16 + lh] = f2bf(h);
            }
        }
        asm volatile("s_wait_dscnt 0x0" ::: "memory");
#pragma unroll 1
        for (int kc2 = 0; kc2 < 4; ++kc2) {  // K = 128 of this ff chunk
            __syncthreads();
            for (int idx = tid; idx < 16 * 256; idx += 128) {
                int kl2 = (idx >> 8) * 2, c = idx & 255;
                float w0 = W2[(size_t)(f * 128 + kc2 * 32 + kl2) * CC + c];
                float w1 = W2[(size_t)(f * 128 + kc2 * 32 + kl2 + 1) * CC + c];
                *(unsigned int*)&wT[c][kl2] = pack2(w0, w1);
            }
            __syncthreads();
            v16bf a = ldsA16(&hb[wave][lh][0], kc2 * 32, lane);
#pragma unroll
            for (int t = 0; t < 16; ++t) {
                v16bf bw = ldsB(&wT[t * 16 + lh][0], 0, lane);
                oacc[t] = wmma_bf16(a, bw, oacc[t]);
            }
        }
    }
    // bias + residual + LN2
#pragma unroll
    for (int t = 0; t < 16; ++t) {
        int c = t * 16 + lh;
        float bb = b2[c];
#pragma unroll
        for (int r = 0; r < 8; ++r) {
            int rowg = base + r + 8 * hi;
            oacc[t][r] += bb + x[(size_t)rowg * CC + c];
        }
    }
    float mu[8], rs[8];
#pragma unroll
    for (int r = 0; r < 8; ++r) {
        float s = 0.f, q = 0.f;
#pragma unroll
        for (int t = 0; t < 16; ++t) { float v = oacc[t][r]; s += v; q += v * v; }
        s = rsum16(s); q = rsum16(q);
        float m = s * (1.0f / 256.0f);
        float var = q * (1.0f / 256.0f) - m * m;
        mu[r] = m; rs[r] = rsqrtf(var + 1e-5f);
    }
#pragma unroll
    for (int t = 0; t < 16; ++t) {
        int c = t * 16 + lh;
        float g = g2[c], bb = be2[c];
#pragma unroll
        for (int r = 0; r < 8; ++r) {
            int rowg = base + r + 8 * hi;
            out[(size_t)rowg * CC + c] = (oacc[t][r] - mu[r]) * rs[r] * g + bb;
        }
    }
}

extern "C" void kernel_launch(void* const* d_in, const int* in_sizes, int n_in,
                              void* d_out, int out_size, void* d_ws, size_t ws_size,
                              hipStream_t stream) {
    (void)in_sizes; (void)n_in; (void)out_size; (void)ws_size;
    const float* src     = (const float*)d_in[0];
    const float* matched = (const float*)d_in[1];
    const float* pos     = (const float*)d_in[2];
    const float* mask    = (const float*)d_in[3];
    const float* Wagg    = (const float*)d_in[4];
    const float* bagg    = (const float*)d_in[5];
    const float* W1      = (const float*)d_in[6];
    const float* b1      = (const float*)d_in[7];
    const float* W2      = (const float*)d_in[8];
    const float* b2      = (const float*)d_in[9];
    const float* g1      = (const float*)d_in[10];
    const float* be1     = (const float*)d_in[11];
    const float* g2      = (const float*)d_in[12];
    const float* be2     = (const float*)d_in[13];
    float* out = (float*)d_out;

    // workspace: cat [b][n][k][c] f32 (32MB) then ln1 [(n,b)][c] f32 (8MB)
    float* ws_att = (float*)d_ws;
    float* ws_ln1 = ws_att + (size_t)BB * NN * KK * CC;

    attn_kernel<<<dim3(512), dim3(128), 0, stream>>>(src, matched, pos, mask, ws_att);
    agg_ln_kernel<<<dim3(128), dim3(128), 0, stream>>>(ws_att, Wagg, bagg, src, g1, be1, ws_ln1);
    ffn_ln_kernel<<<dim3(128), dim3(128), 0, stream>>>(ws_ln1, W1, b1, W2, b2, g2, be2, out);
}
